// TransformerExtendGroup_35648228557128
// MI455X (gfx1250) — matrix operand order repaired and checked by reading, not compiled, not run on hardware
//
#include <hip/hip_runtime.h>
#include <hip/hip_bf16.h>
#include <math.h>

typedef __attribute__((ext_vector_type(16))) _Float16 v16h;
typedef __attribute__((ext_vector_type(8)))  _Float16 v8h;
typedef __attribute__((ext_vector_type(8)))  float    v8f;

#define NPTS 16384
#define NB   4

static __device__ __forceinline__ v8f wmma_f16(v16h a, v16h b, v8f c) {
  // D = A(16x32 f16) * B(32x16 f16) + C(16x16 f32)
  return __builtin_amdgcn_wmma_f32_16x16x32_f16(false, a, false, b, (short)0, c, false, false);
}
static __device__ __forceinline__ v16h hcat(v8h lo, v8h hi) {
  return __builtin_shufflevector(lo, hi, 0,1,2,3,4,5,6,7,8,9,10,11,12,13,14,15);
}
static __device__ __forceinline__ void dsfence() {
  // per-wave LDS ordering: make prior ds stores visible to following ds loads
  asm volatile("s_wait_dscnt 0" ::: "memory");
}
// A fragment (16x32, 16-bit): lane holds row m=lane%16; K = 0..15 (lanes 0-15)
// or 16..31 (lanes 16-31). src = 64+ contiguous f16 of one row, klo = hi?8:0.
static __device__ __forceinline__ v16h fragA64(const _Float16* src, int kc, int klo) {
  const _Float16* p = src + kc * 32 + 2 * klo;
  v8h lo = *(const v8h*)p;
  v8h hi = *(const v8h*)(p + 8);
  return hcat(lo, hi);
}
// B fragment (32x16, 16-bit): lane holds column n=lane%16; K = 0..15 (lanes 0-15),
// 16..31 (lanes 16-31). W stored in LDS as [n_out][kpad] f16, kpad multiple of 16.
static __device__ __forceinline__ v16h fragB(const _Float16* W, int kpad, int nt,
                                             int kc, int lm, int kb) {
  const _Float16* p = W + (nt * 16 + lm) * kpad + kc * 32 + kb;
  return *(const v16h*)p;  // 32B aligned by construction
}

__global__ void __launch_bounds__(128)
pt_attn_kernel(const float* __restrict__ xyz,  const float* __restrict__ nxyz,
               const float* __restrict__ ptsg, const float* __restrict__ npg,
               const float* __restrict__ Wq,   const float* __restrict__ Wk,
               const float* __restrict__ Wv,   const float* __restrict__ Wm1,
               const float* __restrict__ Wm2,  const float* __restrict__ Wpos,
               const float* __restrict__ bpos, float* __restrict__ out)
{
  // ---- weights in B-friendly LDS layout [n_out][k_pad], f16, zero-padded K ----
  __shared__ __align__(32) _Float16 sWv[64 * 160];   // K: 0..63 pts | 64..127 np | 128..130 rel
  __shared__ __align__(32) _Float16 sWk[64 * 64];
  __shared__ __align__(32) _Float16 sWq[64 * 64];
  __shared__ __align__(32) _Float16 sWm1[16 * 64];
  __shared__ __align__(32) _Float16 sWpos[64 * 32];  // K: 0..4 = [feat,xyzn,rel0..2]
  __shared__ float sbpos[64];
  __shared__ float sm2[16];
  // ---- per-wave scratch (4 waves / block) ----
  __shared__ __align__(32) _Float16 sPts[4][64];
  __shared__ __align__(32) _Float16 sNP[4][16 * 64];   // [s][ch]
  __shared__ __align__(32) _Float16 sPos[4][16 * 8];   // [s][feat,xyzn,r0,r1,r2,0,0,0]
  __shared__ __align__(32) _Float16 sStage[4][16 * 64]; // kin -> d -> h(f32) -> partials(f32)
  __shared__ float sLog[4][16];
  __shared__ float sRed[4][16];

  const int tid = threadIdx.x;
  for (int i = tid; i < 64 * 160; i += 128) {
    int n = i / 160, k = i % 160;
    sWv[i] = (k < 131) ? (_Float16)Wv[n * 131 + k] : (_Float16)0.0f;
  }
  for (int i = tid; i < 64 * 64; i += 128) {
    sWk[i] = (_Float16)Wk[i];
    sWq[i] = (_Float16)Wq[i];
  }
  for (int i = tid; i < 16 * 64; i += 128) sWm1[i] = (_Float16)Wm1[i];
  for (int i = tid; i < 64 * 32; i += 128) {
    int n = i >> 5, k = i & 31;
    sWpos[i] = (k < 5) ? (_Float16)Wpos[n * 5 + k] : (_Float16)0.0f;
  }
  if (tid < 64) sbpos[tid] = bpos[tid];
  if (tid < 16) sm2[tid] = Wm2[tid];
  __syncthreads();

  const int wv   = tid >> 5;
  const int lane = tid & 31;
  const int lm   = lane & 15;
  const int hi   = lane >> 4;   // half-wave id
  const int moff = hi * 8;      // C/D row offset
  const int klo  = hi * 8;      // A-fragment K offset
  const int kb   = hi * 16;     // B-fragment K offset

  _Float16* myPts = sPts[wv];
  _Float16* myNP  = sNP[wv];
  _Float16* myPos = sPos[wv];
  _Float16* myStg = sStage[wv];
  float*    hb    = (float*)myStg;

  const long total  = (long)NB * NPTS;
  const long stride = (long)gridDim.x * 4;
  for (long p = (long)blockIdx.x * 4 + wv; p < total; p += stride) {
    const int b = (int)(p >> 14);
    const int n = (int)(p & (NPTS - 1));

    // ---- stage point + neighbor features (f32 -> f16) ----
    myPts[lane]      = (_Float16)ptsg[((size_t)(b * 64 + lane)) * NPTS + n];
    myPts[lane + 32] = (_Float16)ptsg[((size_t)(b * 64 + lane + 32)) * NPTS + n];
    // neighbor tile: 64 ch x 16 s; 16B (4 neighbors) per global_load_b128
    #pragma unroll
    for (int j = 0; j < 8; ++j) {
      int idx4 = lane + j * 32;        // 0..255 quads
      int c  = idx4 >> 2;
      int s0 = (idx4 & 3) * 4;
      float4 vv = *(const float4*)(npg + (((size_t)(b * 64 + c)) * NPTS + n) * 16 + s0);
      myNP[(s0 + 0) * 64 + c] = (_Float16)vv.x;
      myNP[(s0 + 1) * 64 + c] = (_Float16)vv.y;
      myNP[(s0 + 2) * 64 + c] = (_Float16)vv.z;
      myNP[(s0 + 3) * 64 + c] = (_Float16)vv.w;
    }
    // ---- prefetch next point's big streams while we compute this one ----
    {
      long p2 = p + stride;
      if (p2 < total) {
        int b2 = (int)(p2 >> 14);
        int n2 = (int)(p2 & (NPTS - 1));
        const float* npb = npg + (((size_t)(b2 * 64 + lane)) * NPTS + n2) * 16;
        __builtin_prefetch(npb, 0, 1);                               // ch 0..31 lines
        __builtin_prefetch(npb + (size_t)32 * NPTS * 16, 0, 1);      // ch 32..63 lines
        __builtin_prefetch(ptsg + ((size_t)(b2 * 64 + lane)) * NPTS + n2, 0, 1);
        __builtin_prefetch(ptsg + ((size_t)(b2 * 64 + lane + 32)) * NPTS + n2, 0, 1);
      }
    }
    dsfence();

    // ---- rel, feat_norm, xyz_norm (feat_norm split across half-waves) ----
    {
      float fx = 0.0f;
      const int coff = hi * 32;
      #pragma unroll 8
      for (int c = 0; c < 32; ++c) {
        float d = (float)myPts[c + coff] - (float)myNP[lm * 64 + c + coff];
        fx += d * d;
      }
      if (hi) sRed[wv][lm] = fx;
      dsfence();
      if (!hi) {
        fx += sRed[wv][lm];
        float r0 = xyz[((size_t)(b * 3 + 0)) * NPTS + n] -
                   nxyz[(((size_t)(b * 3 + 0)) * NPTS + n) * 16 + lm];
        float r1 = xyz[((size_t)(b * 3 + 1)) * NPTS + n] -
                   nxyz[(((size_t)(b * 3 + 1)) * NPTS + n) * 16 + lm];
        float r2 = xyz[((size_t)(b * 3 + 2)) * NPTS + n] -
                   nxyz[(((size_t)(b * 3 + 2)) * NPTS + n) * 16 + lm];
        myPos[lm * 8 + 0] = (_Float16)sqrtf(fx);
        myPos[lm * 8 + 1] = (_Float16)sqrtf(r0 * r0 + r1 * r1 + r2 * r2);
        myPos[lm * 8 + 2] = (_Float16)r0;
        myPos[lm * 8 + 3] = (_Float16)r1;
        myPos[lm * 8 + 4] = (_Float16)r2;
        myPos[lm * 8 + 5] = (_Float16)0.0f;
        myPos[lm * 8 + 6] = (_Float16)0.0f;
        myPos[lm * 8 + 7] = (_Float16)0.0f;
      }
      dsfence();
    }

    // ---- A fragments (M = 16 neighbors) ----
    v16h aP0 = fragA64(myPts, 0, klo);            // pts broadcast rows, K 0..31
    v16h aP1 = fragA64(myPts, 1, klo);            // K 32..63
    v16h aN0 = fragA64(myNP + lm * 64, 0, klo);   // neighbor feats K 0..31
    v16h aN1 = fragA64(myNP + lm * 64, 1, klo);   // K 32..63
    v16h aPos = {};                                // pos (K 0..4 valid)
    if (!hi) {
      v8h pr = *(const v8h*)(myPos + lm * 8);
      #pragma unroll
      for (int e = 0; e < 8; ++e) aPos[e] = pr[e];
    }
    v16h aRel = {};                                // v_in K 128..130 (rel)
    if (!hi) {
      aRel[0] = myPos[lm * 8 + 2];
      aRel[1] = myPos[lm * 8 + 3];
      aRel[2] = myPos[lm * 8 + 4];
    }

    // ---- v = Wv @ v_in ; kin = v + Wpos@pos + bpos ----
    v8f vt[4];
    #pragma unroll
    for (int nt = 0; nt < 4; ++nt) {
      v8f acc = {};
      acc = wmma_f16(aP0,  fragB(sWv, 160, nt, 0, lm, kb), acc);
      acc = wmma_f16(aP1,  fragB(sWv, 160, nt, 1, lm, kb), acc);
      acc = wmma_f16(aN0,  fragB(sWv, 160, nt, 2, lm, kb), acc);
      acc = wmma_f16(aN1,  fragB(sWv, 160, nt, 3, lm, kb), acc);
      acc = wmma_f16(aRel, fragB(sWv, 160, nt, 4, lm, kb), acc);
      v8f pe = {};
      pe = wmma_f16(aPos, fragB(sWpos, 32, nt, 0, lm, kb), pe);
      float bp = sbpos[nt * 16 + lm];
      #pragma unroll
      for (int j = 0; j < 8; ++j)
        myStg[(j + moff) * 64 + nt * 16 + lm] = (_Float16)(acc[j] + pe[j] + bp);
      vt[nt] = acc;
    }
    dsfence();

    // ---- k = Wk @ kin ; q = Wq @ pts ; stage d = q - k ----
    v16h aK0 = fragA64(myStg + lm * 64, 0, klo);
    v16h aK1 = fragA64(myStg + lm * 64, 1, klo);
    #pragma unroll
    for (int nt = 0; nt < 4; ++nt) {
      v8f kk = {};
      kk = wmma_f16(aK0, fragB(sWk, 64, nt, 0, lm, kb), kk);
      kk = wmma_f16(aK1, fragB(sWk, 64, nt, 1, lm, kb), kk);
      v8f qq = {};
      qq = wmma_f16(aP0, fragB(sWq, 64, nt, 0, lm, kb), qq);
      qq = wmma_f16(aP1, fragB(sWq, 64, nt, 1, lm, kb), qq);
      #pragma unroll
      for (int j = 0; j < 8; ++j)
        myStg[(j + moff) * 64 + nt * 16 + lm] = (_Float16)(qq[j] - kk[j]);
    }
    dsfence();

    // ---- h = relu(Wm1 @ d) ; logits = Wm2 @ h ----
    v16h aD0 = fragA64(myStg + lm * 64, 0, klo);
    v16h aD1 = fragA64(myStg + lm * 64, 1, klo);
    v8f h = {};
    h = wmma_f16(aD0, fragB(sWm1, 64, 0, 0, lm, kb), h);
    h = wmma_f16(aD1, fragB(sWm1, 64, 0, 1, lm, kb), h);
    #pragma unroll
    for (int j = 0; j < 8; ++j) {
      float hv = h[j];
      hb[(j + moff) * 16 + lm] = hv > 0.0f ? hv : 0.0f;
    }
    dsfence();
    if (!hi) {
      float lg = 0.0f;
      #pragma unroll
      for (int nn = 0; nn < 16; ++nn) lg += hb[lm * 16 + nn] * sm2[nn];
      sLog[wv][lm] = lg;
    }
    dsfence();
    if (!hi) {   // softmax over the 16 neighbors
      float mx = -1e30f;
      #pragma unroll
      for (int i = 0; i < 16; ++i) mx = fmaxf(mx, sLog[wv][i]);
      float sum = 0.0f;
      #pragma unroll
      for (int i = 0; i < 16; ++i) sum += __expf(sLog[wv][i] - mx);
      sRed[wv][lm] = __expf(sLog[wv][lm] - mx) / sum;
    }
    dsfence();

    // ---- out[ch] = sum_s w[s] * v[s][ch] ----
    #pragma unroll
    for (int nt = 0; nt < 4; ++nt) {
      float o = 0.0f;
      #pragma unroll
      for (int j = 0; j < 8; ++j) o += sRed[wv][j + moff] * vt[nt][j];
      hb[nt * 32 + lane] = o;   // partial over this half-wave's s range
    }
    dsfence();
    {
      const int t0 = hi * 2;    // lanes 0-15 -> tiles 0,1 ; lanes 16-31 -> tiles 2,3
      #pragma unroll
      for (int t = t0; t < t0 + 2; ++t) {
        float val = hb[t * 32 + lm] + hb[t * 32 + lm + 16];
        out[((size_t)(b * 64 + t * 16 + lm)) * NPTS + n] = val;
      }
    }
  }
}

extern "C" void kernel_launch(void* const* d_in, const int* in_sizes, int n_in,
                              void* d_out, int out_size, void* d_ws, size_t ws_size,
                              hipStream_t stream) {
  const float* xyz  = (const float*)d_in[0];
  const float* nxyz = (const float*)d_in[1];
  const float* pts  = (const float*)d_in[2];
  const float* np   = (const float*)d_in[3];
  const float* Wq   = (const float*)d_in[4];
  const float* Wk   = (const float*)d_in[5];
  const float* Wv   = (const float*)d_in[6];
  const float* Wm1  = (const float*)d_in[7];
  const float* Wm2  = (const float*)d_in[8];
  const float* Wpos = (const float*)d_in[9];
  const float* bp   = (const float*)d_in[10];
  float* out = (float*)d_out;
  (void)in_sizes; (void)n_in; (void)out_size; (void)d_ws; (void)ws_size;
  // 2048 blocks x 4 waves -> 8 points per wave (weight staging amortized)
  pt_attn_kernel<<<dim3(2048), dim3(128), 0, stream>>>(
      xyz, nxyz, pts, np, Wq, Wk, Wv, Wm1, Wm2, Wpos, bp, out);
}
